// LSTMSeqModel_66718021976082
// MI455X (gfx1250) — matrix operand order repaired and checked by reading, not compiled, not run on hardware
//
#include <hip/hip_runtime.h>

// ---------------------------------------------------------------------------
// LSTM (B=32, T=16, D=1024, 4 passes) for gfx1250 using v_wmma_f32_16x16x32_f16
// Software-pipelined GEMM inner loops (register double-buffering).
// ---------------------------------------------------------------------------

typedef __attribute__((ext_vector_type(16))) _Float16 v16h;
typedef __attribute__((ext_vector_type(8)))  _Float16 v8h;
typedef __attribute__((ext_vector_type(8)))  float    v8f;

#define DD     1024
#define FOURD  4096
#define BB     32
#define TT     16
#define NPASS  4
#define HOFF   2097152   // 32*64*1024
#define COFF   2129920   // HOFF + 32768

union AFrag { v16h v; v8h h[2]; };

struct Frags { AFrag a; v16h b0, b1, b2, b3; };

__device__ __forceinline__ v8f wmma16(v16h a, v16h b, v8f c) {
    // 8 args: (neg_a, A, neg_b, B, c_mod, C, reuse_a, reuse_b)
    return __builtin_amdgcn_wmma_f32_16x16x32_f16(false, a, false, b, (short)0, c,
                                                  false, false);
}

__device__ __forceinline__ float sigmoidf_(float x) {
    return 1.0f / (1.0f + __expf(-x));
}

// Load one pipeline stage: A fragment (two b128) + 4 B fragments (two b128 each)
__device__ __forceinline__ void load_frags(Frags& f, const _Float16* __restrict__ arow,
                                           const _Float16* __restrict__ bbase,
                                           int k_t, size_t sstride) {
    f.a.h[0] = *(const v8h*)(arow + k_t * 32);
    f.a.h[1] = *(const v8h*)(arow + k_t * 32 + 16);
    const _Float16* bk = bbase + (size_t)k_t * 512;
    f.b0 = *(const v16h*)(bk);
    f.b1 = *(const v16h*)(bk + sstride);
    f.b2 = *(const v16h*)(bk + 2 * sstride);
    f.b3 = *(const v16h*)(bk + 3 * sstride);
}

__device__ __forceinline__ void mma_frags(const Frags& f, v8f& c0, v8f& c1,
                                          v8f& c2, v8f& c3) {
    c0 = wmma16(f.a.v, f.b0, c0);
    c1 = wmma16(f.a.v, f.b1, c1);
    c2 = wmma16(f.a.v, f.b2, c2);
    c3 = wmma16(f.a.v, f.b3, c3);
}

// ---------------------------------------------------------------------------
// Pack W (row-major [4096,1024] fp32) into f16 WMMA B-fragment order.
// B[k][n] = W[n][k].  Fragment element (n_t,k_t,lane,j):
//   n = n_t*16 + (lane&15),  k = k_t*32 + 8*(lane>>4) + j + 8*(j>>3)   (j = 0..15)
// Each lane's 16 halves are stored contiguously (32B) -> two b128 loads.
// ---------------------------------------------------------------------------
__global__ void __launch_bounds__(256)
pack_weights_kernel(const float* __restrict__ Wih, const float* __restrict__ Whh,
                    _Float16* __restrict__ pkih, _Float16* __restrict__ pkhh) {
    int tid  = blockIdx.x * 256 + threadIdx.x;       // 524288 threads
    int lane = tid & 31;
    int k_t  = (tid >> 5) & 31;
    int n_t  = (tid >> 10) & 255;
    int mat  = tid >> 18;
    const float* src = mat ? Whh : Wih;
    _Float16*    dst = mat ? pkhh : pkih;
    int n  = n_t * 16 + (lane & 15);
    int kb = k_t * 32 + ((lane >> 4) * 8);
    const float* s = src + (size_t)n * DD + kb;
    _Float16*    d = dst + (((size_t)(n_t * 32 + k_t) * 32 + lane) * 16);
#pragma unroll
    for (int j = 0; j < 16; ++j) d[j] = (_Float16)s[j + (j >> 3) * 8];
}

__global__ void __launch_bounds__(256)
bias_kernel(const float* __restrict__ bi, const float* __restrict__ bh,
            float* __restrict__ b) {
    int i = blockIdx.x * 256 + threadIdx.x;
    if (i < FOURD) b[i] = bi[i] + bh[i];
}

__global__ void __launch_bounds__(256)
init_state_kernel(_Float16* __restrict__ h0, float* __restrict__ c0) {
    int i = blockIdx.x * 256 + threadIdx.x;          // 32768
    h0[i] = (_Float16)0.0f;
    c0[i] = 0.0f;
}

__global__ void __launch_bounds__(256)
convert_x_kernel(const float* __restrict__ x, _Float16* __restrict__ xh) {
    int i = blockIdx.x * 256 + threadIdx.x;          // 524288
    xh[i] = (_Float16)x[i];
}

// ---------------------------------------------------------------------------
// xg = x_h[512,1024] @ Wih^T + bias  -> xg[512,4096] fp32
// One wave per 16x64 output tile; 2048 waves = 256 blocks x 8 waves.
// Waves within a block share the same n-group (identical B fragments -> WGP$
// reuse) and cover 8 consecutive m-tiles.
// ---------------------------------------------------------------------------
__global__ void __launch_bounds__(256)
gemm_xg_kernel(const _Float16* __restrict__ xh, const _Float16* __restrict__ Wpk,
               const float* __restrict__ bias, float* __restrict__ xg) {
    int lane = threadIdx.x & 31;
    int wid  = blockIdx.x * 8 + (threadIdx.x >> 5);  // 0..2047
    int m_t  = wid & 31;                             // fast index: shared B
    int n4   = wid >> 5;                             // 0..63
    int m0   = m_t * 16;
    int n0   = n4 * 64;
    int lm   = lane & 15, hi = lane >> 4;

    const _Float16* arow  = xh + (size_t)(m0 + lm) * DD + hi * 8;
    const _Float16* bbase = Wpk + (size_t)(n0 >> 4) * 16384 + lane * 16;

    v8f acc0 = {}, acc1 = {}, acc2 = {}, acc3 = {};

    Frags f0, f1;
    load_frags(f0, arow, bbase, 0, 16384);
#pragma unroll 1
    for (int k = 0; k < 30; k += 2) {
        load_frags(f1, arow, bbase, k + 1, 16384);     // prefetch k+1
        mma_frags(f0, acc0, acc1, acc2, acc3);         // compute k
        load_frags(f0, arow, bbase, k + 2, 16384);     // prefetch k+2
        mma_frags(f1, acc0, acc1, acc2, acc3);         // compute k+1
    }
    load_frags(f1, arow, bbase, 31, 16384);
    mma_frags(f0, acc0, acc1, acc2, acc3);
    mma_frags(f1, acc0, acc1, acc2, acc3);

    v8f accs[4] = {acc0, acc1, acc2, acc3};
#pragma unroll
    for (int s = 0; s < 4; ++s) {
        int n = n0 + s * 16 + lm;
        float bv = bias[n];
#pragma unroll
        for (int v = 0; v < 8; ++v) {
            int m = m0 + v + hi * 8;
            xg[(size_t)m * FOURD + n] = accs[s][v] + bv;
        }
    }
}

// ---------------------------------------------------------------------------
// One recurrent step: gates = xg[:,t,:] + h @ Whh^T ; pointwise LSTM update.
// 128 tile groups (2 m-tiles x 64 d-tiles). Each group = 4 waves (split-K=4),
// each wave computes all 4 gate tiles for its (m,d) patch. Block = 2 groups.
// ---------------------------------------------------------------------------
__global__ void __launch_bounds__(256)
lstm_step_kernel(const _Float16* __restrict__ hprev, const _Float16* __restrict__ Wpk,
                 const float* __restrict__ xg, float* __restrict__ cbuf,
                 _Float16* __restrict__ hnext, _Float16* __restrict__ xh,
                 float* __restrict__ out, int p, int t, int is_last) {
    __shared__ float red[2][3][4][256];              // 24 KB

    int lane = threadIdx.x & 31;
    int w    = threadIdx.x >> 5;                     // 0..7
    int g    = w >> 2;                               // group in block
    int ks   = w & 3;                                // K slice
    int G    = blockIdx.x * 2 + g;                   // 0..127
    int m0   = (G >> 6) * 16;
    int d_t  = G & 63;
    int d0   = d_t * 16;
    int lm   = lane & 15, hi = lane >> 4;

    const _Float16* arow  = hprev + (size_t)(m0 + lm) * DD + hi * 8;
    const _Float16* bbase = Wpk + (size_t)d_t * 16384 + lane * 16;
    int kbase = ks * 8;

    v8f a0 = {}, a1 = {}, a2 = {}, a3 = {};

    Frags f0, f1;
    load_frags(f0, arow, bbase, kbase, 1048576);
#pragma unroll
    for (int i = 0; i < 6; i += 2) {
        load_frags(f1, arow, bbase, kbase + i + 1, 1048576);
        mma_frags(f0, a0, a1, a2, a3);
        load_frags(f0, arow, bbase, kbase + i + 2, 1048576);
        mma_frags(f1, a0, a1, a2, a3);
    }
    load_frags(f1, arow, bbase, kbase + 7, 1048576);
    mma_frags(f0, a0, a1, a2, a3);
    mma_frags(f1, a0, a1, a2, a3);

    v8f accq[4] = {a0, a1, a2, a3};
    if (ks != 0) {
#pragma unroll
        for (int q = 0; q < 4; ++q)
#pragma unroll
            for (int v = 0; v < 8; ++v)
                red[g][ks - 1][q][lane * 8 + v] = accq[q][v];
    }
    __syncthreads();

    if (ks == 0) {
        int d = d0 + lm;
#pragma unroll
        for (int v = 0; v < 8; ++v) {
            int m = m0 + v + hi * 8;
            size_t xrow = ((size_t)m * TT + t) * FOURD;
            float gt[4];
#pragma unroll
            for (int q = 0; q < 4; ++q) {
                int li = lane * 8 + v;
                gt[q] = accq[q][v] + red[g][0][q][li] + red[g][1][q][li] +
                        red[g][2][q][li] + xg[xrow + (size_t)q * DD + d];
            }
            float iv = sigmoidf_(gt[0]);
            float fv = sigmoidf_(gt[1]);
            float gv = tanhf(gt[2]);
            float ov = sigmoidf_(gt[3]);
            size_t cidx = (size_t)m * DD + d;
            float cn = fv * cbuf[cidx] + iv * gv;
            float hn = ov * tanhf(cn);
            cbuf[cidx]  = cn;
            hnext[cidx] = (_Float16)hn;
            xh[((size_t)m * TT + t) * DD + d] = (_Float16)hn;
            __builtin_nontemporal_store(
                hn, &out[((size_t)m * (NPASS * TT) + p * TT + t) * DD + d]);
            if (is_last) {
                out[HOFF + cidx] = hn;
                out[COFF + cidx] = cn;
            }
        }
    }
}

// ---------------------------------------------------------------------------
// Host launcher
// ---------------------------------------------------------------------------
extern "C" void kernel_launch(void* const* d_in, const int* in_sizes, int n_in,
                              void* d_out, int out_size, void* d_ws, size_t ws_size,
                              hipStream_t stream) {
    (void)in_sizes; (void)n_in; (void)out_size; (void)ws_size;
    const float* x   = (const float*)d_in[0];
    const float* Wih = (const float*)d_in[1];
    const float* Whh = (const float*)d_in[2];
    const float* bih = (const float*)d_in[3];
    const float* bhh = (const float*)d_in[4];
    float* out = (float*)d_out;
    char*  ws  = (char*)d_ws;

    _Float16* pkih  = (_Float16*)(ws + 0);          //  8,388,608 B
    _Float16* pkhh  = (_Float16*)(ws + 8388608);    //  8,388,608 B
    float*    bias  = (float*)   (ws + 16777216);   //     16,384 B
    float*    xg    = (float*)   (ws + 16793600);   //  8,388,608 B
    _Float16* xh    = (_Float16*)(ws + 25182208);   //  1,048,576 B
    _Float16* hbuf0 = (_Float16*)(ws + 26230784);   //     65,536 B
    _Float16* hbuf1 = (_Float16*)(ws + 26296320);   //     65,536 B
    float*    cbuf  = (float*)   (ws + 26361856);   //    131,072 B

    pack_weights_kernel<<<2048, 256, 0, stream>>>(Wih, Whh, pkih, pkhh);
    bias_kernel<<<16, 256, 0, stream>>>(bih, bhh, bias);
    init_state_kernel<<<128, 256, 0, stream>>>(hbuf0, cbuf);
    convert_x_kernel<<<2048, 256, 0, stream>>>(x, xh);

    for (int p = 0; p < NPASS; ++p) {
        gemm_xg_kernel<<<256, 256, 0, stream>>>(xh, pkih, bias, xg);
        for (int t = 0; t < TT; ++t) {
            int s = p * TT + t;
            const _Float16* hr = (s & 1) ? hbuf1 : hbuf0;
            _Float16*       hw = (s & 1) ? hbuf0 : hbuf1;
            int last = (p == NPASS - 1 && t == TT - 1) ? 1 : 0;
            lstm_step_kernel<<<64, 256, 0, stream>>>(hr, pkhh, xg, cbuf, hw, xh,
                                                     out, p, t, last);
        }
    }
}
